// DGCNN_76759655514521
// MI455X (gfx1250) — compile-verified, matrix operand order checked by reading, not run on hardware
//
#include <hip/hip_runtime.h>
#include <hip/hip_bf16.h>

// ---------------------------------------------------------------------------
// DGCNN feature extractor for MI455X (gfx1250), wave32 + WMMA bf16.
//   B=16, C=3, N=2048, K=20.  P = B*N*K = 655360 GEMM rows for conv1..4.
// Intermediates stored bf16 (halves HBM traffic, the binding resource).
// Weight tiles staged to LDS with global_load_async_to_lds_b128 (ASYNCcnt).
// ---------------------------------------------------------------------------

#define KNN 20
#define BB  16
#define NN  2048
static constexpr long long PP = (long long)BB * NN * KNN;   // 655360
static constexpr int M5 = BB * NN;                          // 32768

typedef __attribute__((ext_vector_type(16))) __bf16 v16bf;
typedef __attribute__((ext_vector_type(8)))  float  v8f;

__device__ __forceinline__ unsigned short f2bf(float f) {
  unsigned u = __float_as_uint(f);
  u += 0x7FFFu + ((u >> 16) & 1u);        // round-to-nearest-even
  return (unsigned short)(u >> 16);
}
__device__ __forceinline__ float bf2f(unsigned short h) {
  return __uint_as_float(((unsigned)h) << 16);
}

union FragU { uint4 q[2]; v16bf v; };

// Fragment gather matching the CDNA5 16-bit A/B layout:
// lane L: row/col = L&15, khalf = L>>4; VGPR v<4 -> K = 2v + 8*khalf,
// v>=4 -> K = 16 + 2(v-4) + 8*khalf.  Tile stored [row][32] bf16 in LDS,
// so each fragment = two contiguous 16B LDS loads.
__device__ __forceinline__ v16bf ldfrag(const unsigned short* t, int row, int khalf) {
  FragU f;
  f.q[0] = *reinterpret_cast<const uint4*>(t + row * 32 + khalf * 8);
  f.q[1] = *reinterpret_cast<const uint4*>(t + row * 32 + 16 + khalf * 8);
  return f.v;
}

// ---------------------------------------------------------------------------
// Kernel 1: kNN (top-20 by -dist, self included, stable ties) + edge features
// e[pos][8] = {xj-xi (3), xi (3), 0, 0},  pos = (b*N + i)*K + k   (fp32)
// ---------------------------------------------------------------------------
__global__ __launch_bounds__(256) void knn_edge_kernel(const float* __restrict__ x,
                                                       float* __restrict__ e) {
  __shared__ float sx0[NN], sx1[NN], sx2[NN], sxx[NN];
  const int tid = threadIdx.x;
  const int b = blockIdx.y;
  const float* xb = x + (size_t)b * 3 * NN;
  for (int i = tid; i < NN; i += 256) {
    sx0[i] = xb[i]; sx1[i] = xb[NN + i]; sx2[i] = xb[2 * NN + i];
  }
  __syncthreads();
  for (int i = tid; i < NN; i += 256)
    sxx[i] = sx0[i] * sx0[i] + sx1[i] * sx1[i] + sx2[i] * sx2[i];
  __syncthreads();

  const int i = blockIdx.x * 256 + tid;
  const float xi0 = sx0[i], xi1 = sx1[i], xi2 = sx2[i], xxi = sxx[i];

  float vals[KNN]; int idxs[KNN];
#pragma unroll
  for (int k = 0; k < KNN; ++k) { vals[k] = -3.4e38f; idxs[k] = 0; }

  for (int j = 0; j < NN; ++j) {
    float d = 2.f * (xi0 * sx0[j] + xi1 * sx1[j] + xi2 * sx2[j]) - xxi - sxx[j];
    if (d > vals[KNN - 1]) {                 // stable sorted insert (registers)
      float cv = d; int ci = j;
#pragma unroll
      for (int p = 0; p < KNN; ++p) {
        bool take = cv > vals[p];
        float tv = vals[p]; int ti = idxs[p];
        vals[p] = take ? cv : tv; idxs[p] = take ? ci : ti;
        cv = take ? tv : cv;      ci = take ? ti : ci;
      }
    }
  }

  const size_t pos2 = (size_t)b * NN + i;
  float* ep = e + pos2 * KNN * 8;
#pragma unroll
  for (int k = 0; k < KNN; ++k) {
    int j = idxs[k];
    float* p = ep + k * 8;
    p[0] = sx0[j] - xi0; p[1] = sx1[j] - xi1; p[2] = sx2[j] - xi2;
    p[3] = xi0; p[4] = xi1; p[5] = xi2; p[6] = 0.f; p[7] = 0.f;
  }
}

// ---------------------------------------------------------------------------
// Pre-convert weights to bf16, zero-padded [cout][cinp] (== WMMA Bt layout)
// ---------------------------------------------------------------------------
__global__ __launch_bounds__(256) void convert_w_kernel(const float* __restrict__ w,
                                                        unsigned short* __restrict__ o,
                                                        int cin, int cinp, int total) {
  int idx = blockIdx.x * 256 + threadIdx.x;
  if (idx >= total) return;
  int r = idx / cinp, c = idx % cinp;
  o[idx] = f2bf(c < cin ? w[r * cin + c] : 0.f);
}

// ---------------------------------------------------------------------------
// Generic fused GEMM kernel: y = W * transform(in)
//   - in  : [M][CSTRIDE] (bf16 if IN_BF16 else fp32), pre-BN of prev layer
//   - TRANSFORM: apply (sc,sh) affine + ReLU of the *previous* BN in staging
//   - B (weights, bf16, pre-padded) staged via global_load_async_to_lds_b128
//   - WMMA bf16 16x16x32, K chunked by 32; stats fused into epilogue
// Block: 128 threads = 4 waves; block tile 64(M) x 64(N); grid (M/64, COUT/64)
// ---------------------------------------------------------------------------
template <int CIN, int CSTRIDE, int COUT, bool TRANSFORM, bool IN_BF16>
__global__ __launch_bounds__(128) void conv_wmma_kernel(
    const void* __restrict__ in_, const unsigned short* __restrict__ wbf,
    const float* __restrict__ sc, const float* __restrict__ sh,
    unsigned short* __restrict__ out, float* __restrict__ gsum, float* __restrict__ gsq) {
  __shared__ __align__(16) unsigned short aT[64 * 32];   // [pos][k] bf16
  __shared__ __align__(16) unsigned short bT[64 * 32];   // [n][k]   bf16
  __shared__ float ssum[64], ssq[64];

  const float* inf = (const float*)in_;
  const unsigned short* inb = (const unsigned short*)in_;

  const int tid = threadIdx.x;
  const int lane = tid & 31;
  const int wv = tid >> 5;
  const int mBase = blockIdx.x * 64;
  const int nBase = blockIdx.y * 64;
  if (tid < 64) { ssum[tid] = 0.f; ssq[tid] = 0.f; }

  v8f acc[4] = {};
  const int col = lane & 15;
  const int khalf = lane >> 4;
  constexpr int NCH = CIN / 32;
  const unsigned bT_lds = (unsigned)(unsigned long long)(void*)bT;  // LDS offset
  const char* wrow = (const char*)(wbf + (size_t)nBase * CIN);      // block's 64 rows

  for (int kc = 0; kc < NCH; ++kc) {
    // --- B tile: 64 rows x 32 k bf16 = 4KB, raw async copy to LDS ----------
#pragma unroll
    for (int h = 0; h < 2; ++h) {                       // 256 x 16B transfers
      int idx = tid + h * 128;
      int row = idx >> 2, q = idx & 3;
      unsigned long long ga = (unsigned long long)(const char*)(
          wrow + (size_t)row * CIN * 2 + (size_t)kc * 64 + q * 16);
      unsigned la = bT_lds + (unsigned)(row * 64 + q * 16);
      asm volatile("global_load_async_to_lds_b128 %0, %1, off"
                   :: "v"(la), "v"(ga) : "memory");
    }
    // --- A tile: 64 pos x 32 cin, fused BN+ReLU+bf16 convert ---------------
#pragma unroll
    for (int it = 0; it < 16; ++it) {
      int eI = tid + it * 128;
      int r = eI >> 5, c = eI & 31;
      int cg = kc * 32 + c;
      float v = 0.f;
      if (cg < CSTRIDE) {
        if constexpr (IN_BF16) v = bf2f(inb[(size_t)(mBase + r) * CSTRIDE + cg]);
        else                   v = inf[(size_t)(mBase + r) * CSTRIDE + cg];
        if constexpr (TRANSFORM) {
          v = v * sc[cg] + sh[cg];
          v = v > 0.f ? v : 0.f;
        }
      }
      aT[eI] = f2bf(v);
    }
    if (kc + 1 < NCH) {                     // hint next K-chunk (global_prefetch_b8)
      const size_t eb = IN_BF16 ? 2 : 4;
      __builtin_prefetch((const char*)in_ +
                             ((size_t)(mBase + (tid >> 1)) * CSTRIDE +
                              (size_t)(kc + 1) * 32) * eb + (size_t)(tid & 1) * 16 * eb,
                         0, 3);
    }
    asm volatile("s_wait_asynccnt 0x0" ::: "memory");   // own wave's B transfers done
    __syncthreads();                                    // all waves' tiles visible

    v16bf a = ldfrag(aT, wv * 16 + col, khalf);
#pragma unroll
    for (int t = 0; t < 4; ++t) {
      v16bf bfrag = ldfrag(bT, t * 16 + col, khalf);
      acc[t] = __builtin_amdgcn_wmma_f32_16x16x32_bf16(
          false, a, false, bfrag, (short)0, acc[t], false, false);
    }
    __syncthreads();
  }

  // Epilogue: store pre-BN y as bf16, accumulate BN stats (fp32) per channel.
#pragma unroll
  for (int t = 0; t < 4; ++t) {
    int n = nBase + t * 16 + col;
    float s = 0.f, s2 = 0.f;
#pragma unroll
    for (int v = 0; v < 8; ++v) {           // D layout: row = v + 8*khalf
      float d = acc[t][v];
      int row = mBase + wv * 16 + khalf * 8 + v;
      out[(size_t)row * COUT + n] = f2bf(d);
      s += d; s2 += d * d;
    }
    atomicAdd(&ssum[t * 16 + col], s);      // ds_add_f32
    atomicAdd(&ssq[t * 16 + col], s2);
  }
  __syncthreads();
  if (tid < 64) {
    atomicAdd(&gsum[nBase + tid], ssum[tid]);
    atomicAdd(&gsq[nBase + tid], ssq[tid]);
  }
}

// ---------------------------------------------------------------------------
// BN finalize: stats -> per-channel affine (scale, shift)
// ---------------------------------------------------------------------------
__global__ void bn_finalize_kernel(const float* __restrict__ gsum, const float* __restrict__ gsq,
                                   const float* __restrict__ g, const float* __restrict__ b,
                                   float* __restrict__ sc, float* __restrict__ sh, float invCnt) {
  int c = threadIdx.x;
  float mean = gsum[c] * invCnt;
  float var = gsq[c] * invCnt - mean * mean;
  float is = rsqrtf(var + 1e-5f);
  float s = g[c] * is;
  sc[c] = s;
  sh[c] = b[c] - mean * s;
}

// ---------------------------------------------------------------------------
// Max over K of relu(y*sc+sh) -> xcat (bf16) channel slice [coff, coff+C)
// ---------------------------------------------------------------------------
__global__ __launch_bounds__(256) void maxpool_kernel(
    const unsigned short* __restrict__ y, const float* __restrict__ sc,
    const float* __restrict__ sh, unsigned short* __restrict__ xcat, int C, int coff) {
  long long idx = (long long)blockIdx.x * 256 + threadIdx.x;
  long long total = (long long)M5 * C;
  if (idx >= total) return;
  int c = (int)(idx % C);
  long long pos2 = idx / C;
  float s = sc[c], h0 = sh[c];
  float m = -3.4e38f;
  const unsigned short* p = y + (size_t)pos2 * KNN * C + c;
#pragma unroll
  for (int k = 0; k < KNN; ++k) {
    float h = bf2f(p[(size_t)k * C]) * s + h0;
    h = h > 0.f ? h : 0.f;
    m = m > h ? m : h;
  }
  xcat[(size_t)pos2 * 512 + coff + c] = f2bf(m);
}

// ---------------------------------------------------------------------------
// Final: out[b][c][n] = relu(bn5(y5))   ([pos][c] bf16 -> [b,c,n] fp32)
// ---------------------------------------------------------------------------
__global__ __launch_bounds__(256) void final_kernel(
    const unsigned short* __restrict__ y5, const float* __restrict__ sc,
    const float* __restrict__ sh, float* __restrict__ out) {
  long long idx = (long long)blockIdx.x * 256 + threadIdx.x;
  if (idx >= (long long)BB * 512 * NN) return;
  int n = (int)(idx % NN);
  int c = (int)((idx / NN) % 512);
  int b = (int)(idx / ((long long)512 * NN));
  float v = bf2f(y5[((size_t)b * NN + n) * 512 + c]) * sc[c] + sh[c];
  out[idx] = v > 0.f ? v : 0.f;
}

// ---------------------------------------------------------------------------
extern "C" void kernel_launch(void* const* d_in, const int* in_sizes, int n_in,
                              void* d_out, int out_size, void* d_ws, size_t ws_size,
                              hipStream_t stream) {
  (void)in_sizes; (void)n_in; (void)out_size; (void)ws_size;
  const float* x  = (const float*)d_in[0];
  const float* w1 = (const float*)d_in[1];  const float* g1 = (const float*)d_in[2];  const float* b1 = (const float*)d_in[3];
  const float* w2 = (const float*)d_in[4];  const float* g2 = (const float*)d_in[5];  const float* b2 = (const float*)d_in[6];
  const float* w3 = (const float*)d_in[7];  const float* g3 = (const float*)d_in[8];  const float* b3 = (const float*)d_in[9];
  const float* w4 = (const float*)d_in[10]; const float* g4 = (const float*)d_in[11]; const float* b4 = (const float*)d_in[12];
  const float* w5 = (const float*)d_in[13]; const float* g5 = (const float*)d_in[14]; const float* b5 = (const float*)d_in[15];
  float* out = (float*)d_out;

  char* wsb = (char*)d_ws;
  float* ws = (float*)d_ws;
  // BN stats (zeroed each launch) + affine params: 4096 floats (16KB)
  float* s1 = ws;          float* q1 = s1 + 64;
  float* s2 = q1 + 64;     float* q2 = s2 + 64;
  float* s3 = q2 + 64;     float* q3 = s3 + 128;
  float* s4 = q3 + 128;    float* q4 = s4 + 256;
  float* s5 = q4 + 256;    float* q5 = s5 + 512;
  float* sc1 = ws + 2048;  float* sh1 = sc1 + 64;
  float* sc2 = sh1 + 64;   float* sh2 = sc2 + 64;
  float* sc3 = sh2 + 64;   float* sh3 = sc3 + 128;
  float* sc4 = sh3 + 128;  float* sh4 = sc4 + 256;
  float* sc5 = sh4 + 256;  float* sh5 = sc5 + 512;
  // bf16 weights, padded [cout][cinp]
  unsigned short* wb1 = (unsigned short*)(wsb + 16384);   // 64 x 32
  unsigned short* wb2 = wb1 + 64 * 32;                    // 64 x 64
  unsigned short* wb3 = wb2 + 64 * 64;                    // 128 x 64
  unsigned short* wb4 = wb3 + 128 * 64;                   // 256 x 128
  unsigned short* wb5 = wb4 + 256 * 128;                  // 512 x 512
  // Big buffers (e fp32; activations bf16)
  float*          e    = (float*)(wsb + 634880);          // [P][8] fp32
  unsigned short* xcat = (unsigned short*)(wsb + 634880 + (size_t)PP * 8 * 4);
  unsigned short* y5   = xcat + (size_t)M5 * 512;
  unsigned short* y1   = y5   + (size_t)M5 * 512;
  unsigned short* y2   = y1   + (size_t)PP * 64;
  unsigned short* y3   = y2   + (size_t)PP * 64;
  unsigned short* y4   = y3   + (size_t)PP * 128;

  hipMemsetAsync(ws, 0, 2048 * sizeof(float), stream);    // stats only

  const float invP  = 1.f / (float)PP;
  const float invP5 = 1.f / (float)M5;
  const unsigned gM  = (unsigned)(PP / 64);   // 10240
  const unsigned gM5 = (unsigned)(M5 / 64);   // 512

  // 0) weights -> bf16 padded
  convert_w_kernel<<<(64 * 32 + 255) / 256, 256, 0, stream>>>(w1, wb1, 6, 32, 64 * 32);
  convert_w_kernel<<<(64 * 64 + 255) / 256, 256, 0, stream>>>(w2, wb2, 64, 64, 64 * 64);
  convert_w_kernel<<<(128 * 64 + 255) / 256, 256, 0, stream>>>(w3, wb3, 64, 64, 128 * 64);
  convert_w_kernel<<<(256 * 128 + 255) / 256, 256, 0, stream>>>(w4, wb4, 128, 128, 256 * 128);
  convert_w_kernel<<<(512 * 512 + 255) / 256, 256, 0, stream>>>(w5, wb5, 512, 512, 512 * 512);

  // 1) kNN + edge features
  knn_edge_kernel<<<dim3(NN / 256, BB), 256, 0, stream>>>(x, e);

  // 2) conv1 (6 -> 64, K padded 6->32)
  conv_wmma_kernel<32, 8, 64, false, false><<<dim3(gM, 1), 128, 0, stream>>>(
      e, wb1, nullptr, nullptr, y1, s1, q1);
  bn_finalize_kernel<<<1, 64, 0, stream>>>(s1, q1, g1, b1, sc1, sh1, invP);
  maxpool_kernel<<<(unsigned)(((long long)M5 * 64 + 255) / 256), 256, 0, stream>>>(
      y1, sc1, sh1, xcat, 64, 0);

  // 3) conv2 (64 -> 64), BN1+ReLU fused into staging
  conv_wmma_kernel<64, 64, 64, true, true><<<dim3(gM, 1), 128, 0, stream>>>(
      y1, wb2, sc1, sh1, y2, s2, q2);
  bn_finalize_kernel<<<1, 64, 0, stream>>>(s2, q2, g2, b2, sc2, sh2, invP);
  maxpool_kernel<<<(unsigned)(((long long)M5 * 64 + 255) / 256), 256, 0, stream>>>(
      y2, sc2, sh2, xcat, 64, 64);

  // 4) conv3 (64 -> 128)
  conv_wmma_kernel<64, 64, 128, true, true><<<dim3(gM, 2), 128, 0, stream>>>(
      y2, wb3, sc2, sh2, y3, s3, q3);
  bn_finalize_kernel<<<1, 128, 0, stream>>>(s3, q3, g3, b3, sc3, sh3, invP);
  maxpool_kernel<<<(unsigned)(((long long)M5 * 128 + 255) / 256), 256, 0, stream>>>(
      y3, sc3, sh3, xcat, 128, 128);

  // 5) conv4 (128 -> 256)
  conv_wmma_kernel<128, 128, 256, true, true><<<dim3(gM, 4), 128, 0, stream>>>(
      y3, wb4, sc3, sh3, y4, s4, q4);
  bn_finalize_kernel<<<1, 256, 0, stream>>>(s4, q4, g4, b4, sc4, sh4, invP);
  maxpool_kernel<<<(unsigned)(((long long)M5 * 256 + 255) / 256), 256, 0, stream>>>(
      y4, sc4, sh4, xcat, 256, 256);

  // 6) conv5 (512 -> 512) on concat of maxes (already post-activation)
  conv_wmma_kernel<512, 512, 512, false, true><<<dim3(gM5, 8), 128, 0, stream>>>(
      xcat, wb5, nullptr, nullptr, y5, s5, q5);
  bn_finalize_kernel<<<1, 512, 0, stream>>>(s5, q5, g5, b5, sc5, sh5, invP5);

  // 7) final BN + ReLU -> [B, 512, N] fp32
  final_kernel<<<(unsigned)(((long long)BB * 512 * NN + 255) / 256), 256, 0, stream>>>(
      y5, sc5, sh5, out);
}
